// Decoder_1743756722637
// MI455X (gfx1250) — compile-verified, matrix operand order checked by reading, not run on hardware
//
#include <hip/hip_runtime.h>
#include <hip/hip_bf16.h>
#include <math.h>

// ---------------------------------------------------------------------------
// Problem constants: B,L,H,E,R,T = 256,100,1000,100,247,15
// ---------------------------------------------------------------------------
#define BB   256
#define LL   100
#define HH   1000
#define EE   100
#define RR   247
#define TT   15
#define G3   3000            // 3*H
#define FF   100             // fuse dim
#define NOUT 349             // R+1 + L+1

// padded dims (K padded to 32 for WMMA chunks, N padded to 64 = 16*NT)
#define KP_H 1024            // ceil32(1000)
#define KP_E 128             // ceil32(100)
#define NP_G3 3008           // ceil64(3000)
#define NP_R  256            // ceil64(247)
#define NP_F  128            // ceil64(100)
#define NP_E  128            // ceil64(100)
#define NT    4              // N tiles (of 16) per wave

typedef __attribute__((ext_vector_type(16))) __bf16 v16bf;
typedef __attribute__((ext_vector_type(8)))  __bf16 v8bf;
typedef __attribute__((ext_vector_type(8)))  float  v8f;

// ---------------------------------------------------------------------------
// device helpers
// ---------------------------------------------------------------------------
__device__ __forceinline__ float selu_f(float x) {
    const float a = 1.6732632423543772f, s = 1.0507009873554805f;
    return s * (x > 0.f ? x : a * (__expf(x) - 1.f));
}
__device__ __forceinline__ float sigmoid_f(float x) {
    return 1.f / (1.f + __expf(-x));
}
__device__ __forceinline__ float wave_sum(float v) {
    #pragma unroll
    for (int off = 16; off > 0; off >>= 1) v += __shfl_xor(v, off, 32);
    return v;
}
__device__ __forceinline__ float wave_max(float v) {
    #pragma unroll
    for (int off = 16; off > 0; off >>= 1) v = fmaxf(v, __shfl_xor(v, off, 32));
    return v;
}
// Build a 16-element bf16 fragment from two contiguous 16-byte runs.
// Fragment K packing (CDNA5 ISA): elems 0..7 <- p[0..7], 8..15 <- p[16..23].
__device__ __forceinline__ v16bf make_frag(const __bf16* __restrict__ p) {
    v8bf lo = *(const v8bf*)p;
    v8bf hi = *(const v8bf*)(p + 16);
    v16bf r;
    #pragma unroll
    for (int i = 0; i < 8; ++i) { r[i] = lo[i]; r[i + 8] = hi[i]; }
    return r;
}

// ---------------------------------------------------------------------------
// WMMA GEMM: C[M,N] = A[M,K] @ B[K,N] (+ bias[N])
//   A : bf16 [M, ldak], row stride ldak = Kpad (zero-padded K)
//   Bt: bf16 [Npad, ldbk] = B^T with zero-padded rows/cols
//   One wave computes a 16 x 64 C strip (4 WMMA tiles, A fragment reused).
//   Software double-buffered K loop: chunk k+1 fragment loads are issued
//   before the 4 WMMAs of chunk k, so VMEM latency overlaps matrix ops.
// ---------------------------------------------------------------------------
__global__ __launch_bounds__(32) void wmma_gemm_bf16(
    const __bf16* __restrict__ A, int ldak,
    const __bf16* __restrict__ Bt, int ldbk,
    const float* __restrict__ bias,
    float* __restrict__ C, int ldc,
    int N, int K)
{
    const int lane = threadIdx.x & 31;
    const int half = lane >> 4;          // 0: lanes 0-15, 1: lanes 16-31
    const int lr   = lane & 15;
    const int tm   = blockIdx.y * 16;
    const int tn0  = blockIdx.x * (16 * NT);

    const __bf16* pa  = A  + (size_t)(tm + lr) * ldak + half * 8;
    const __bf16* pb0 = Bt + (size_t)(tn0 +  0 + lr) * ldbk + half * 8;
    const __bf16* pb1 = Bt + (size_t)(tn0 + 16 + lr) * ldbk + half * 8;
    const __bf16* pb2 = Bt + (size_t)(tn0 + 32 + lr) * ldbk + half * 8;
    const __bf16* pb3 = Bt + (size_t)(tn0 + 48 + lr) * ldbk + half * 8;

    v8f acc0 = {}, acc1 = {}, acc2 = {}, acc3 = {};

    // prologue: load chunk 0
    v16bf a  = make_frag(pa);
    v16bf b0 = make_frag(pb0);
    v16bf b1 = make_frag(pb1);
    v16bf b2 = make_frag(pb2);
    v16bf b3 = make_frag(pb3);

    const int nk = (K + 31) >> 5;
    for (int kc = 0; kc < nk - 1; ++kc) {
        pa += 32; pb0 += 32; pb1 += 32; pb2 += 32; pb3 += 32;
        // issue next chunk's loads before consuming current fragments
        const v16bf an  = make_frag(pa);
        const v16bf bn0 = make_frag(pb0);
        const v16bf bn1 = make_frag(pb1);
        const v16bf bn2 = make_frag(pb2);
        const v16bf bn3 = make_frag(pb3);
        acc0 = __builtin_amdgcn_wmma_f32_16x16x32_bf16(false, a, false, b0, (short)0, acc0, false, false);
        acc1 = __builtin_amdgcn_wmma_f32_16x16x32_bf16(false, a, false, b1, (short)0, acc1, false, false);
        acc2 = __builtin_amdgcn_wmma_f32_16x16x32_bf16(false, a, false, b2, (short)0, acc2, false, false);
        acc3 = __builtin_amdgcn_wmma_f32_16x16x32_bf16(false, a, false, b3, (short)0, acc3, false, false);
        a = an; b0 = bn0; b1 = bn1; b2 = bn2; b3 = bn3;
    }
    // epilogue: last chunk
    acc0 = __builtin_amdgcn_wmma_f32_16x16x32_bf16(false, a, false, b0, (short)0, acc0, false, false);
    acc1 = __builtin_amdgcn_wmma_f32_16x16x32_bf16(false, a, false, b1, (short)0, acc1, false, false);
    acc2 = __builtin_amdgcn_wmma_f32_16x16x32_bf16(false, a, false, b2, (short)0, acc2, false, false);
    acc3 = __builtin_amdgcn_wmma_f32_16x16x32_bf16(false, a, false, b3, (short)0, acc3, false, false);

    v8f accs[NT] = {acc0, acc1, acc2, acc3};
    #pragma unroll
    for (int t = 0; t < NT; ++t) {
        const int col = tn0 + t * 16 + lr;
        if (col < N) {
            const float bb = bias ? bias[col] : 0.f;
            #pragma unroll
            for (int i = 0; i < 8; ++i)      // C layout: lanes 16-31 -> rows M+8..15
                C[(size_t)(tm + half * 8 + i) * ldc + col] = accs[t][i] + bb;
        }
    }
}

// out[row*stride] = dot(A[row, 0:K], w) + bias   (one wave per row)
__global__ __launch_bounds__(32) void rowdot(
    const float* __restrict__ A, int lda, int M, int K,
    const float* __restrict__ w, const float* __restrict__ bias,
    float* __restrict__ out, int out_stride)
{
    const int row = blockIdx.x;
    if (row >= M) return;
    const int lane = threadIdx.x;
    float acc = 0.f;
    for (int k = lane; k < K; k += 32)
        acc += A[(size_t)row * lda + k] * w[k];
    acc = wave_sum(acc);
    if (lane == 0) out[(size_t)row * out_stride] = acc + (bias ? bias[0] : 0.f);
}

// in-place row softmax (one wave per row)
__global__ __launch_bounds__(32) void softmax_rows(float* __restrict__ s, int n)
{
    const int r = blockIdx.x, lane = threadIdx.x;
    float* row = s + (size_t)r * n;
    float m = -3.402823466e38f;
    for (int k = lane; k < n; k += 32) m = fmaxf(m, row[k]);
    m = wave_max(m);
    float sum = 0.f;
    for (int k = lane; k < n; k += 32) sum += __expf(row[k] - m);
    sum = wave_sum(sum);
    const float inv = 1.f / sum;
    for (int k = lane; k < n; k += 32) row[k] = __expf(row[k] - m) * inv;
}

// c[b,h] = sum_n wts[b,n] * mem[b,n,h]
__global__ void attn_ctx(const float* __restrict__ wts,
                         const float* __restrict__ mem,
                         float* __restrict__ c, int Bn, int N, int Hd)
{
    const int idx = blockIdx.x * blockDim.x + threadIdx.x;
    if (idx >= Bn * Hd) return;
    const int b = idx / Hd, hcol = idx % Hd;
    const float* mb = mem + (size_t)b * N * Hd + hcol;
    const float* wb = wts + (size_t)b * N;
    float acc = 0.f;
    for (int n = 0; n < N; ++n) acc += wb[n] * mb[(size_t)n * Hd];
    c[idx] = acc;
}

// transpose + convert weight: src f32 [K,N] row-major  ->  dst bf16 [Npad, Kpad]
__global__ void cvt_w_t(const float* __restrict__ src, __bf16* __restrict__ dst,
                        int N, int K, int kpad)
{
    const int idx = blockIdx.x * blockDim.x + threadIdx.x;
    if (idx >= N * K) return;
    const int n = idx / K, k = idx % K;
    dst[(size_t)n * kpad + k] = (__bf16)src[(size_t)k * N + n];
}

// convert rows f32 -> bf16 with stride change and optional SELU
__global__ void cvt_rows(const float* __restrict__ src, int ld_src,
                         __bf16* __restrict__ dst, int ld_dst,
                         int rows, int K, int do_selu)
{
    const int idx = blockIdx.x * blockDim.x + threadIdx.x;
    if (idx >= rows * K) return;
    const int r = idx / K, k = idx % K;
    float v = src[(size_t)r * ld_src + k];
    if (do_selu) v = selu_f(v);
    dst[(size_t)r * ld_dst + k] = (__bf16)v;
}

// x_bf[b,j] = bf16( xe[(b*T+t), j] + xc[b, j] )
__global__ void build_x(const float* __restrict__ xe, const float* __restrict__ xc,
                        __bf16* __restrict__ xbf, int t)
{
    const int idx = blockIdx.x * blockDim.x + threadIdx.x;
    if (idx >= BB * EE) return;
    const int b = idx / EE, j = idx % EE;
    xbf[(size_t)b * KP_E + j] = (__bf16)(xe[((size_t)b * TT + t) * EE + j] + xc[idx]);
}

// GRU gates; writes h (f32), h_bf, hs_bf = bf16(selu(h))
__global__ void gru_update(const float* __restrict__ gi, const float* __restrict__ gh,
                           float* __restrict__ h,
                           __bf16* __restrict__ h_bf, __bf16* __restrict__ hs_bf)
{
    const int idx = blockIdx.x * blockDim.x + threadIdx.x;
    if (idx >= BB * HH) return;
    const int b = idx / HH, j = idx % HH;
    const float* gib = gi + (size_t)b * G3;
    const float* ghb = gh + (size_t)b * G3;
    const float r = sigmoid_f(gib[j] + ghb[j]);
    const float z = sigmoid_f(gib[HH + j] + ghb[HH + j]);
    const float n = tanhf(gib[2 * HH + j] + r * ghb[2 * HH + j]);
    const float hn = (1.f - z) * n + z * h[idx];
    h[idx] = hn;
    h_bf[(size_t)b * KP_H + j]  = (__bf16)hn;
    hs_bf[(size_t)b * KP_H + j] = (__bf16)selu_f(hn);
}

// cp[b,l] = dot(selu(hf[b,:] + enc_fuse[b,l,:] + b_fuse), W_copy) + b_copy
__global__ __launch_bounds__(32) void copy_scores(
    const float* __restrict__ hf, const float* __restrict__ enc_fuse,
    const float* __restrict__ b_fuse, const float* __restrict__ W_copy,
    const float* __restrict__ b_copy, float* __restrict__ cp)
{
    const int i = blockIdx.x;                  // b*L + l
    if (i >= BB * LL) return;
    const int b = i / LL, l = i % LL;
    const int lane = threadIdx.x;
    float acc = 0.f;
    for (int k = lane; k < FF; k += 32) {
        const float v = hf[(size_t)b * FF + k] + enc_fuse[(size_t)i * FF + k] + b_fuse[k];
        acc += selu_f(v) * W_copy[k];
    }
    acc = wave_sum(acc);
    if (lane == 0) cp[(size_t)b * (LL + 1) + l] = acc + b_copy[0];
}

// out[r, k] = log_softmax(in[r, :n])[k]   (one wave per row)
__global__ __launch_bounds__(32) void logsoftmax_rows(
    const float* __restrict__ in, int n, int ld_in,
    float* __restrict__ out, int ld_out)
{
    const int r = blockIdx.x, lane = threadIdx.x;
    const float* row = in + (size_t)r * ld_in;
    float m = -3.402823466e38f;
    for (int k = lane; k < n; k += 32) m = fmaxf(m, row[k]);
    m = wave_max(m);
    float s = 0.f;
    for (int k = lane; k < n; k += 32) s += __expf(row[k] - m);
    s = wave_sum(s);
    const float ls = __logf(s);
    float* orow = out + (size_t)r * ld_out;
    for (int k = lane; k < n; k += 32) orow[k] = row[k] - m - ls;
}

// ---------------------------------------------------------------------------
// launch
// ---------------------------------------------------------------------------
extern "C" void kernel_launch(void* const* d_in, const int* in_sizes, int n_in,
                              void* d_out, int out_size, void* d_ws, size_t ws_size,
                              hipStream_t stream)
{
    const float* emb_seq = (const float*)d_in[0];   // [B,T,E]
    const float* enc     = (const float*)d_in[1];   // [B,L,H]
    const float* conv    = (const float*)d_in[2];   // [B,R,H]
    const float* h0      = (const float*)d_in[3];   // [B,H]
    const float* W_attn  = (const float*)d_in[4];   // [2H,1]
    const float* b_attn  = (const float*)d_in[5];
    const float* W_comb  = (const float*)d_in[6];   // [E+H,E]
    const float* b_comb  = (const float*)d_in[7];
    const float* W_ih    = (const float*)d_in[8];   // [E,3H]
    const float* W_hh    = (const float*)d_in[9];   // [H,3H]
    const float* b_ih    = (const float*)d_in[10];
    const float* b_hh    = (const float*)d_in[11];
    const float* W_eos   = (const float*)d_in[12];  // [H,1]
    const float* b_eos   = (const float*)d_in[13];
    const float* W_pred  = (const float*)d_in[14];  // [H,R]
    const float* b_pred  = (const float*)d_in[15];
    const float* W_fuse  = (const float*)d_in[16];  // [2H,100]
    const float* b_fuse  = (const float*)d_in[17];
    const float* W_copy  = (const float*)d_in[18];  // [100,1]
    const float* b_copy  = (const float*)d_in[19];
    float* out = (float*)d_out;                      // [T,B,349]
    (void)in_sizes; (void)n_in; (void)out_size; (void)ws_size;

    // ---- bump allocator over d_ws ----
    char* wsB = (char*)d_ws;
    size_t off = 0;
    auto alloc = [&](size_t bytes) -> void* {
        void* p = wsB + off;
        off = (off + bytes + 255) & ~(size_t)255;
        return p;
    };

    // bf16 arena (zeroed once per launch so padded lanes contribute 0)
    const size_t bf_start = off;
    __bf16* Wt_hh    = (__bf16*)alloc((size_t)NP_G3 * KP_H * 2);
    __bf16* Wt_ih    = (__bf16*)alloc((size_t)NP_G3 * KP_E * 2);
    __bf16* Wt_pred  = (__bf16*)alloc((size_t)NP_R  * KP_H * 2);
    __bf16* Wt_fuseT = (__bf16*)alloc((size_t)NP_F  * KP_H * 2);
    __bf16* Wt_fuseB = (__bf16*)alloc((size_t)NP_F  * KP_H * 2);
    __bf16* Wt_combT = (__bf16*)alloc((size_t)NP_E  * KP_E * 2);
    __bf16* Wt_combB = (__bf16*)alloc((size_t)NP_E  * KP_H * 2);
    __bf16* encS_bf  = (__bf16*)alloc((size_t)BB * LL * KP_H * 2); // selu(enc)
    __bf16* cEnc_bf  = (__bf16*)alloc((size_t)BB * KP_H * 2);
    __bf16* rcConv_bf= (__bf16*)alloc((size_t)BB * KP_H * 2);
    __bf16* emb_bf   = (__bf16*)alloc((size_t)BB * TT * KP_E * 2);
    __bf16* h_bf     = (__bf16*)alloc((size_t)BB * KP_H * 2);
    __bf16* hs_bf    = (__bf16*)alloc((size_t)BB * KP_H * 2);
    __bf16* x_bf     = (__bf16*)alloc((size_t)BB * KP_E * 2);
    const size_t bf_bytes = off - bf_start;

    float* enc_score  = (float*)alloc((size_t)BB * LL * 4);
    float* conv_score = (float*)alloc((size_t)BB * RR * 4);
    float* enc_fuse   = (float*)alloc((size_t)BB * LL * FF * 4);
    float* c_enc      = (float*)alloc((size_t)BB * HH * 4);
    float* rc_conv    = (float*)alloc((size_t)BB * HH * 4);
    float* xc_enc     = (float*)alloc((size_t)BB * EE * 4);
    float* xc_conv    = (float*)alloc((size_t)BB * EE * 4);
    float* xe         = (float*)alloc((size_t)BB * TT * EE * 4);
    float* h          = (float*)alloc((size_t)BB * HH * 4);
    float* gi         = (float*)alloc((size_t)BB * G3 * 4);
    float* gh         = (float*)alloc((size_t)BB * G3 * 4);
    float* hf         = (float*)alloc((size_t)BB * FF * 4);
    float* predbuf    = (float*)alloc((size_t)BB * (RR + 1) * 4);
    float* cpbuf      = (float*)alloc((size_t)BB * (LL + 1) * 4);

    hipMemsetAsync(wsB + bf_start, 0, bf_bytes, stream);

    auto gemm = [&](const __bf16* A, int ldak, const __bf16* Bt, int ldbk,
                    const float* bias, float* C, int ldc, int M, int N, int K) {
        dim3 grid((N + 16 * NT - 1) / (16 * NT), M / 16, 1);
        wmma_gemm_bf16<<<grid, dim3(32, 1, 1), 0, stream>>>(
            A, ldak, Bt, ldbk, bias, C, ldc, N, K);
    };
    auto wcvt = [&](const float* s, __bf16* d, int N, int K, int kpad) {
        cvt_w_t<<<((N * K) + 255) / 256, 256, 0, stream>>>(s, d, N, K, kpad);
    };
    auto rcvt = [&](const float* s, int lds, __bf16* d, int ldd, int rows, int K, int selu) {
        cvt_rows<<<((rows * K) + 255) / 256, 256, 0, stream>>>(s, lds, d, ldd, rows, K, selu);
    };

    // ---- one-time weight transpose+convert to bf16 (L2-resident thereafter) ----
    wcvt(W_hh,   Wt_hh,   G3, HH, KP_H);
    wcvt(W_ih,   Wt_ih,   G3, EE, KP_E);
    wcvt(W_pred, Wt_pred, RR, HH, KP_H);
    wcvt(W_fuse,                     Wt_fuseT, FF, HH, KP_H);   // W_fuse[:H]
    wcvt(W_fuse + (size_t)HH * FF,   Wt_fuseB, FF, HH, KP_H);   // W_fuse[H:]
    wcvt(W_comb,                     Wt_combT, EE, EE, KP_E);   // W_comb[:E]
    wcvt(W_comb + (size_t)EE * EE,   Wt_combB, EE, HH, KP_H);   // W_comb[E:]

    // ---- activations -> bf16 ----
    rcvt(enc,     HH, encS_bf, KP_H, BB * LL, HH, /*selu*/1);
    rcvt(emb_seq, EE, emb_bf,  KP_E, BB * TT, EE, 0);
    rcvt(h0,      HH, h_bf,    KP_H, BB,      HH, 0);

    // ---- attention: softmax over per-row-constant-shifted scores is
    //      independent of h => contexts are time-invariant ----
    rowdot<<<BB * LL, 32, 0, stream>>>(enc, HH, BB * LL, HH, W_attn + HH, b_attn, enc_score, 1);
    rowdot<<<BB * RR, 32, 0, stream>>>(conv, HH, BB * RR, HH, W_attn + HH, b_attn, conv_score, 1);
    softmax_rows<<<BB, 32, 0, stream>>>(enc_score, LL);
    softmax_rows<<<BB, 32, 0, stream>>>(conv_score, RR);
    {
        const int n = BB * HH;
        attn_ctx<<<(n + 255) / 256, 256, 0, stream>>>(enc_score,  enc,  c_enc,   BB, LL, HH);
        attn_ctx<<<(n + 255) / 256, 256, 0, stream>>>(conv_score, conv, rc_conv, BB, RR, HH);
    }
    rcvt(c_enc,   HH, cEnc_bf,   KP_H, BB, HH, 0);
    rcvt(rc_conv, HH, rcConv_bf, KP_H, BB, HH, 0);

    // enc_fuse = selu(enc) @ W_fuse[H:]          [B*L,100]
    gemm(encS_bf, KP_H, Wt_fuseB, KP_H, nullptr, enc_fuse, FF, BB * LL, FF, HH);
    // xc_* = c @ W_comb[E:] + b_comb             [B,100]
    gemm(cEnc_bf,   KP_H, Wt_combB, KP_H, b_comb, xc_enc,  EE, BB, EE, HH);
    gemm(rcConv_bf, KP_H, Wt_combB, KP_H, b_comb, xc_conv, EE, BB, EE, HH);
    // xe = emb_seq @ W_comb[:E]                  [B*T,100]
    gemm(emb_bf, KP_E, Wt_combT, KP_E, nullptr, xe, EE, BB * TT, EE, EE);

    // ---- h = h0 ----
    hipMemcpyAsync(h, h0, (size_t)BB * HH * 4, hipMemcpyDeviceToDevice, stream);

    // ---- recurrence ----
    for (int t = 0; t < TT; ++t) {
        const float* xcsel = (t % 3 == 0) ? xc_conv : xc_enc;
        build_x<<<(BB * EE + 255) / 256, 256, 0, stream>>>(xe, xcsel, x_bf, t);

        gemm(x_bf, KP_E, Wt_ih, KP_E, b_ih, gi, G3, BB, G3, EE);   // gi = x@W_ih+b
        gemm(h_bf, KP_H, Wt_hh, KP_H, b_hh, gh, G3, BB, G3, HH);   // gh = h@W_hh+b
        gru_update<<<(BB * HH + 255) / 256, 256, 0, stream>>>(gi, gh, h, h_bf, hs_bf);

        // pred logits into predbuf[:,0:247]; eos into col 247 and cpbuf col 100
        gemm(h_bf, KP_H, Wt_pred, KP_H, b_pred, predbuf, RR + 1, BB, RR, HH);
        rowdot<<<BB, 32, 0, stream>>>(h, HH, BB, HH, W_eos, b_eos, predbuf + RR, RR + 1);
        rowdot<<<BB, 32, 0, stream>>>(h, HH, BB, HH, W_eos, b_eos, cpbuf + LL,  LL + 1);

        // hf = selu(h) @ W_fuse[:H]
        gemm(hs_bf, KP_H, Wt_fuseT, KP_H, nullptr, hf, FF, BB, FF, HH);
        copy_scores<<<BB * LL, 32, 0, stream>>>(hf, enc_fuse, b_fuse, W_copy, b_copy, cpbuf);

        float* outt = out + (size_t)t * BB * NOUT;
        logsoftmax_rows<<<BB, 32, 0, stream>>>(predbuf, RR + 1, RR + 1, outt, NOUT);
        logsoftmax_rows<<<BB, 32, 0, stream>>>(cpbuf,   LL + 1, LL + 1, outt + (RR + 1), NOUT);
    }
}
